// ResidualVQ_19602230739353
// MI455X (gfx1250) — matrix operand order repaired and checked by reading, not compiled, not run on hardware
//
#include <hip/hip_runtime.h>

// ---------------------------------------------------------------------------
// ResidualVQ forward for MI455X (gfx1250, wave32, WMMA).
// z: [16, 256, 4096] f32   codebooks: [8, 1024, 256] f32
// Outputs (flat, in order): quantized_out [16,256,4096], indices [8,16,4096]
//                           (as float), zeros[8], zeros[8],
//                           all_quantized [8,16,256,4096]
// argmin(dist) == argmax(residual_row . normalized_code)
//   -> bf16 WMMA GEMM [65536x256]x[256x1024] + per-row argmax, per quantizer.
// Each wave owns TWO 16-row tiles so every B fragment read from LDS feeds
// two v_wmma_f32_16x16x32_bf16 (halves LDS traffic per WMMA, two
// independent accumulator chains for the XDL pipe).
// ---------------------------------------------------------------------------

typedef __bf16 bf16;
typedef __attribute__((ext_vector_type(16))) __bf16 v16bf;
typedef __attribute__((ext_vector_type(8)))  float  v8f;
typedef __attribute__((ext_vector_type(4)))  __bf16 v4bf;

#define BQ 16
#define DQ 256
#define TQ 4096
#define KQ 1024
#define NQ 8
#define BT (BQ * TQ)          // 65536 rows
#define LDSROW 264            // 256 bf16 + 16B pad -> conflict-free ds_load_b128

union V16 { v16bf v; uint4 q[2]; };

// ---------------------------------------------------------------------------
// 1) Normalize codebooks to bf16 (one wave per code row).
// ---------------------------------------------------------------------------
__global__ void __launch_bounds__(256) rvq_norm_cb(const float* __restrict__ cb,
                                                   bf16* __restrict__ CN) {
    const int row  = blockIdx.x * 8 + (threadIdx.x >> 5);   // 8192 rows
    const int lane = threadIdx.x & 31;
    const float* src = cb + (size_t)row * DQ;
    float vals[8];
    float ss = 0.f;
#pragma unroll
    for (int j = 0; j < 8; ++j) { float x = src[lane + j * 32]; vals[j] = x; ss += x * x; }
#pragma unroll
    for (int m = 16; m >= 1; m >>= 1) ss += __shfl_xor(ss, m, 32);
    const float scale = 1.0f / fmaxf(sqrtf(ss), 1e-12f);
    bf16* dst = CN + (size_t)row * DQ;
#pragma unroll
    for (int j = 0; j < 8; ++j) dst[lane + j * 32] = (bf16)(vals[j] * scale);
}

// ---------------------------------------------------------------------------
// 2) LDS-tiled transpose: z[b,d,t] -> R[(b*T+t), d] f32 and Rb bf16.
//    grid = (T/32, D/32, B), block = 256 (=32x8).
// ---------------------------------------------------------------------------
__global__ void __launch_bounds__(256) rvq_transpose_in(const float* __restrict__ z,
                                                        float* __restrict__ R,
                                                        bf16* __restrict__ Rb) {
    __shared__ float tile[32][33];
    const int b  = blockIdx.z;
    const int d0 = blockIdx.y * 32;
    const int t0 = blockIdx.x * 32;
    const int tx = threadIdx.x & 31;
    const int ty = threadIdx.x >> 5;

    const float* zp = z + ((size_t)b * DQ + d0) * TQ + t0;
#pragma unroll
    for (int j = 0; j < 4; ++j) {
        const int d = ty + j * 8;
        tile[d][tx] = zp[(size_t)d * TQ + tx];
    }
    __syncthreads();

    float* rp  = R  + ((size_t)b * TQ + t0) * DQ + d0;
    bf16*  rbp = Rb + ((size_t)b * TQ + t0) * DQ + d0;
#pragma unroll
    for (int j = 0; j < 4; ++j) {
        const int t = ty + j * 8;
        const float v = tile[tx][t];
        rp [(size_t)t * DQ + tx] = v;
        rbp[(size_t)t * DQ + tx] = (bf16)v;
    }
}

// ---------------------------------------------------------------------------
// 3) WMMA argmax: scores[65536 x 1024] via v_wmma_f32_16x16x32_bf16.
//    block = 256 = 8 waves; each wave owns 2 x 16 rows -> 256 rows per WG.
//    grid = 256. Codebook chunks (16 codes x 256 dims bf16) double-buffered
//    in padded LDS; each B fragment feeds both row tiles.
// ---------------------------------------------------------------------------
__global__ void __launch_bounds__(256) rvq_argmax(const bf16* __restrict__ Rb,
                                                  const bf16* __restrict__ CNn,
                                                  int* __restrict__ idx_out) {
    __shared__ bf16 bsh[2][16 * LDSROW];        // 2 x ~8.25 KB
    const int tid     = threadIdx.x;
    const int lane    = tid & 31;
    const int wave    = tid >> 5;
    const int lrow    = lane & 15;
    const int halfsel = lane >> 4;
    const int rowBase = blockIdx.x * 256 + wave * 32;

    // --- load A fragments (2 tiles x 16 rows x 256 dims bf16) into VGPRs ---
    // 16-bit A layout: lanes 0-15 hold K {8k..8k+7, 8k+16..8k+23} per k-step,
    // lanes 16-31 the complementary K groups.
    V16 afrag0[8], afrag1[8];
    const bf16* arow0 = Rb + (size_t)(rowBase + lrow) * DQ;
    const bf16* arow1 = arow0 + (size_t)16 * DQ;
#pragma unroll
    for (int ks = 0; ks < 8; ++ks) {
        const int off = ks * 32 + halfsel * 8;
        afrag0[ks].q[0] = *(const uint4*)(arow0 + off);
        afrag0[ks].q[1] = *(const uint4*)(arow0 + off + 16);
        afrag1[ks].q[0] = *(const uint4*)(arow1 + off);
        afrag1[ks].q[1] = *(const uint4*)(arow1 + off + 16);
    }

    float bestv0[8], bestv1[8];
    int   besti0[8], besti1[8];
#pragma unroll
    for (int v = 0; v < 8; ++v) {
        bestv0[v] = -3.4e38f; besti0[v] = 0;
        bestv1[v] = -3.4e38f; besti1[v] = 0;
    }

    const int cload = tid >> 4;       // code row this thread stages
    const int sload = tid & 15;       // 16-element segment

    for (int chunk = 0; chunk < 64; ++chunk) {
        bf16* dst = &bsh[chunk & 1][0];
        const bf16* src = CNn + (size_t)chunk * 16 * DQ;
        // cooperative stage: 16 codes x 256 bf16 (each thread 32 B)
        *(uint4*)(dst + cload * LDSROW + sload * 16)     =
            *(const uint4*)(src + cload * DQ + sload * 16);
        *(uint4*)(dst + cload * LDSROW + sload * 16 + 8) =
            *(const uint4*)(src + cload * DQ + sload * 16 + 8);
        if (chunk < 62) {   // hide L2 latency of the staging two chunks ahead
            __builtin_prefetch(src + (size_t)2 * 16 * DQ + cload * DQ + sload * 16, 0, 1);
        }
        __syncthreads();    // double buffer -> single barrier per chunk

        v8f c0 = {0.f, 0.f, 0.f, 0.f, 0.f, 0.f, 0.f, 0.f};
        v8f c1 = {0.f, 0.f, 0.f, 0.f, 0.f, 0.f, 0.f, 0.f};
#pragma unroll
        for (int ks = 0; ks < 8; ++ks) {
            // B layout (32x16 bf16): lanes 0-15 hold K {0..7, 16..23}, lanes 16-31 K {8..15, 24..31}
            V16 bfrag;
            const bf16* bp = dst + lrow * LDSROW + ks * 32 + halfsel * 8;
            bfrag.q[0] = *(const uint4*)(bp);
            bfrag.q[1] = *(const uint4*)(bp + 16);
            c0 = __builtin_amdgcn_wmma_f32_16x16x32_bf16(
                     false, afrag0[ks].v, false, bfrag.v, (short)0, c0, false, false);
            c1 = __builtin_amdgcn_wmma_f32_16x16x32_bf16(
                     false, afrag1[ks].v, false, bfrag.v, (short)0, c1, false, false);
        }

        const int nb = chunk * 16 + lrow;   // this lane's n-track
#pragma unroll
        for (int v = 0; v < 8; ++v) {
            const float s0 = c0[v];
            if (s0 > bestv0[v]) { bestv0[v] = s0; besti0[v] = nb; }
            const float s1 = c1[v];
            if (s1 > bestv1[v]) { bestv1[v] = s1; besti1[v] = nb; }
        }
    }

    // reduce across the 16 n-track lanes (XOR masks stay within each half)
#pragma unroll
    for (int v = 0; v < 8; ++v) {
        float bv0 = bestv0[v]; int bi0 = besti0[v];
        float bv1 = bestv1[v]; int bi1 = besti1[v];
#pragma unroll
        for (int m = 8; m >= 1; m >>= 1) {
            const float ov0 = __shfl_xor(bv0, m, 32);
            const int   oi0 = __shfl_xor(bi0, m, 32);
            if (ov0 > bv0 || (ov0 == bv0 && oi0 < bi0)) { bv0 = ov0; bi0 = oi0; }
            const float ov1 = __shfl_xor(bv1, m, 32);
            const int   oi1 = __shfl_xor(bi1, m, 32);
            if (ov1 > bv1 || (ov1 == bv1 && oi1 < bi1)) { bv1 = ov1; bi1 = oi1; }
        }
        if (lrow == 0) {
            idx_out[rowBase +      halfsel * 8 + v] = bi0;
            idx_out[rowBase + 16 + halfsel * 8 + v] = bi1;
        }
    }
}

// ---------------------------------------------------------------------------
// 4) Residual update in [bt][d] layout (fully coalesced) + index output.
//    grid = 16384, block = 256 (4 rows/block, 4 dims/thread).
// ---------------------------------------------------------------------------
__global__ void __launch_bounds__(256) rvq_update_residual(const float* __restrict__ cb_n,
                                                           const int* __restrict__ idx,
                                                           float* __restrict__ R,
                                                           bf16* __restrict__ Rb,
                                                           float* __restrict__ out_idx_n) {
    const int tid = threadIdx.x;
    const int bt  = blockIdx.x * 4 + (tid >> 6);
    const int d4  = (tid & 63) * 4;
    const int k   = idx[bt];
    if ((tid & 63) == 0) out_idx_n[bt] = (float)k;

    const float4 c = *(const float4*)(cb_n + (size_t)k * DQ + d4);
    float4 r = *(float4*)(R + (size_t)bt * DQ + d4);
    r.x -= c.x; r.y -= c.y; r.z -= c.z; r.w -= c.w;
    *(float4*)(R + (size_t)bt * DQ + d4) = r;
    v4bf rb = { (bf16)r.x, (bf16)r.y, (bf16)r.z, (bf16)r.w };
    *(v4bf*)(Rb + (size_t)bt * DQ + d4) = rb;
}

// ---------------------------------------------------------------------------
// 5) Decode into output layout [b, d, t] (coalesced f32x4 stores; codebook
//    gathers are L2-resident: 1 MB per quantizer).
// ---------------------------------------------------------------------------
__global__ void __launch_bounds__(256) rvq_write_quantized(const float* __restrict__ cb_n,
                                                           const int* __restrict__ idx,
                                                           float* __restrict__ out_all_n) {
    const size_t gid = (size_t)blockIdx.x * 256 + threadIdx.x;  // 16*256*1024
    const int t4 = (int)(gid & 1023) * 4;
    const int d  = (int)(gid >> 10) & 255;
    const int b  = (int)(gid >> 18);
    const int* ip = idx + (size_t)b * TQ + t4;
    float4 o;
    o.x = cb_n[(size_t)ip[0] * DQ + d];
    o.y = cb_n[(size_t)ip[1] * DQ + d];
    o.z = cb_n[(size_t)ip[2] * DQ + d];
    o.w = cb_n[(size_t)ip[3] * DQ + d];
    *(float4*)(out_all_n + (((size_t)b * DQ + d) * TQ + t4)) = o;
}

// ---------------------------------------------------------------------------
// 6) quantized_out = sum_n all_quantized[n]; also write the 16 zero losses.
// ---------------------------------------------------------------------------
__global__ void __launch_bounds__(256) rvq_finalize(const float* __restrict__ out_all,
                                                    float* __restrict__ out_q,
                                                    float* __restrict__ zeros16) {
    const size_t gid = (size_t)blockIdx.x * 256 + threadIdx.x;
    const size_t off = gid * 4;
    float4 s = {0.f, 0.f, 0.f, 0.f};
#pragma unroll
    for (int n = 0; n < NQ; ++n) {
        const float4 a = *(const float4*)(out_all + (size_t)n * (BQ * DQ * TQ) + off);
        s.x += a.x; s.y += a.y; s.z += a.z; s.w += a.w;
    }
    *(float4*)(out_q + off) = s;
    if (gid < 16) zeros16[gid] = 0.f;
}

// ---------------------------------------------------------------------------
extern "C" void kernel_launch(void* const* d_in, const int* in_sizes, int n_in,
                              void* d_out, int out_size, void* d_ws, size_t ws_size,
                              hipStream_t stream) {
    (void)in_sizes; (void)n_in; (void)out_size; (void)ws_size;

    const float* z  = (const float*)d_in[0];   // [16, 256, 4096]
    const float* cb = (const float*)d_in[1];   // [8, 1024, 256]

    float* out      = (float*)d_out;
    float* out_q    = out;                                   // 16*256*4096
    float* out_idx  = out_q   + (size_t)BQ * DQ * TQ;        // 8*16*4096
    float* out_z    = out_idx + (size_t)NQ * BQ * TQ;        // 8 + 8 zeros
    float* out_all  = out_z   + 16;                          // 8*16*256*4096

    char* w = (char*)d_ws;
    float* R  = (float*)w;                 w += (size_t)BT * DQ * sizeof(float);     // 64 MB
    bf16*  Rb = (bf16*)w;                  w += (size_t)BT * DQ * sizeof(bf16);      // 32 MB
    bf16*  CN = (bf16*)w;                  w += (size_t)NQ * KQ * DQ * sizeof(bf16); // 4 MB
    int*   IDX = (int*)w;                                                            // 256 KB

    // one-time prep
    rvq_norm_cb<<<KQ * NQ / 8, 256, 0, stream>>>(cb, CN);
    dim3 tg(TQ / 32, DQ / 32, BQ);
    rvq_transpose_in<<<tg, 256, 0, stream>>>(z, R, Rb);

    // 8 sequential quantizers
    for (int n = 0; n < NQ; ++n) {
        const float* cb_n = cb + (size_t)n * KQ * DQ;
        rvq_argmax<<<BT / 256, 256, 0, stream>>>(Rb, CN + (size_t)n * KQ * DQ, IDX);
        rvq_update_residual<<<BT / 4, 256, 0, stream>>>(cb_n, IDX, R, Rb,
                                                        out_idx + (size_t)n * BT);
        rvq_write_quantized<<<(BQ * DQ * TQ / 4) / 256, 256, 0, stream>>>(
            cb_n, IDX, out_all + (size_t)n * BQ * DQ * TQ);
    }

    rvq_finalize<<<(BQ * DQ * TQ / 4) / 256, 256, 0, stream>>>(out_all, out_q, out_z);
}
